// ConvKNRM_52922587021474
// MI455X (gfx1250) — matrix operand order repaired and checked by reading, not compiled, not run on hardware
//
#include <hip/hip_runtime.h>
#include <hip/hip_bf16.h>

typedef __attribute__((ext_vector_type(16))) __bf16 v16bf;
typedef __attribute__((ext_vector_type(8)))  __bf16 v8bf;
typedef __attribute__((ext_vector_type(8)))  float  v8f;
typedef int v4i_vs __attribute__((vector_size(16)));   // matches builtin param

#define NB   256   // batch
#define LQ   32
#define LD   1024
#define EMB  128
#define NF   128
#define NKK  11

#if __has_builtin(__builtin_amdgcn_global_load_async_to_lds_b128)
#define HAVE_ASYNC_LDS 1
#endif

__device__ __forceinline__ void wait_async0() {
#if __has_builtin(__builtin_amdgcn_s_wait_asynccnt)
  __builtin_amdgcn_s_wait_asynccnt(0);
#else
  asm volatile("s_wait_asynccnt 0x0" ::: "memory");
#endif
}

// ---------------- cross-lane butterfly via ds_swizzle (group-of-32 mode) --
// offset = xor_mask<<10 | or_mask<<5 | and_mask ; and=0x1F keeps full wave
__device__ __forceinline__ float reduce_half16(float x) {
  // sum over the 16 lanes of each half-wave (xor 1,2,4,8 stay in-half)
  x += __int_as_float(__builtin_amdgcn_ds_swizzle(__float_as_int(x), 0x041F));
  x += __int_as_float(__builtin_amdgcn_ds_swizzle(__float_as_int(x), 0x081F));
  x += __int_as_float(__builtin_amdgcn_ds_swizzle(__float_as_int(x), 0x101F));
  x += __int_as_float(__builtin_amdgcn_ds_swizzle(__float_as_int(x), 0x201F));
  return x;
}
__device__ __forceinline__ float reduce_wave32(float x) {
  x = reduce_half16(x);
  x += __int_as_float(__builtin_amdgcn_ds_swizzle(__float_as_int(x), 0x401F));
  return x;
}

__device__ __forceinline__ float fast_exp2(float x) {
#if __has_builtin(__builtin_amdgcn_exp2f)
  return __builtin_amdgcn_exp2f(x);
#else
  return exp2f(x);
#endif
}

// ---------------- WMMA fragment helpers (bf16 16x16x32) ----------------
// A matrix 16x32 (MxK), row-major source, leading dim ld (elements).
// ISA layout: lanes 0-15 -> M=lane, elems[0:8)=K 0..7, elems[8:16)=K 16..23
//             lanes 16-31 -> M=lane-16, elems = K 8..15 and K 24..31
__device__ __forceinline__ v16bf frag_a_rm(const __bf16* p0, int ld) {
  const int lane = threadIdx.x & 31;
  const int m = lane & 15;
  const int g = lane >> 4;
  const __bf16* r = p0 + m * ld;
  v8bf lo = *reinterpret_cast<const v8bf*>(r + g * 8);
  v8bf hi = *reinterpret_cast<const v8bf*>(r + 16 + g * 8);
  v16bf o;
#pragma unroll
  for (int i = 0; i < 8; ++i) { o[i] = lo[i]; o[8 + i] = hi[i]; }
  return o;
}

// B matrix 32x16 (KxN): source stored with each N-column contiguous in K,
// element (k,n) at p0[n*ldcol + k]. Lane n<16 holds K 0..7,16..23 of column n,
// lanes 16-31 hold K 8..15,24..31 of column n-16.
__device__ __forceinline__ v16bf frag_b_kc(const __bf16* p0, int ldcol) {
  const int lane = threadIdx.x & 31;
  const int n = lane & 15;
  const int g = lane >> 4;
  const __bf16* c = p0 + n * ldcol + g * 8;
  v8bf lo = *reinterpret_cast<const v8bf*>(c);
  v8bf hi = *reinterpret_cast<const v8bf*>(c + 16);
  v16bf o;
#pragma unroll
  for (int i = 0; i < 8; ++i) { o[i] = lo[i]; o[8 + i] = hi[i]; }
  return o;
}

__device__ __forceinline__ v8f wmma_bf16(v16bf a, v16bf b, v8f c) {
  return __builtin_amdgcn_wmma_f32_16x16x32_bf16(false, a, false, b,
                                                 (short)0, c, false, false);
}

__device__ __forceinline__ v8f v8f_zero() {
  v8f z = {0.f, 0.f, 0.f, 0.f, 0.f, 0.f, 0.f, 0.f};
  return z;
}

// ---------------- zero workspace ----------------
__global__ __launch_bounds__(256)
void zero_f32_kernel(float* p, int n) {
  int i = blockIdx.x * blockDim.x + threadIdx.x;
  if (i < n) p[i] = 0.f;
}

// ---------------- weight repack: (NF,EMB,n) f32 -> k-major bf16 slices ----
// s=0 -> (n=1,k=0); s=1,2 -> (n=2,k=0,1); s=3,4,5 -> (n=3,k=0..2)
__global__ __launch_bounds__(256)
void prep_weights_kernel(const float* __restrict__ w1,
                         const float* __restrict__ w2,
                         const float* __restrict__ w3,
                         __bf16* __restrict__ Wout) {
  int idx = blockIdx.x * blockDim.x + threadIdx.x;
  if (idx >= 6 * NF * EMB) return;
  int s = idx / (NF * EMB);
  int rem = idx - s * (NF * EMB);
  int f = rem / EMB;
  int c = rem - f * EMB;
  int n, k;
  const float* w;
  if (s == 0)      { n = 1; k = 0;     w = w1; }
  else if (s <= 2) { n = 2; k = s - 1; w = w2; }
  else             { n = 3; k = s - 3; w = w3; }
  Wout[idx] = (__bf16)w[((size_t)f * EMB + c) * n + k];
}

// ---------------- embed + L2 normalize -> bf16, token-major, zero pad -----
// Eout: (NB, Lpad, EMB) bf16; rows [L, Lpad) are zeros (conv right-padding).
__global__ __launch_bounds__(256)
void embed_kernel(const int* __restrict__ ids, const float* __restrict__ emb,
                  __bf16* __restrict__ Eout, int L, int Lpad, int ntok) {
  int wid  = (int)((blockIdx.x * (size_t)blockDim.x + threadIdx.x) >> 5);
  int lane = threadIdx.x & 31;
  if (wid >= ntok) return;           // wave-uniform (wid identical across wave)
  int b = wid / Lpad;
  int t = wid - b * Lpad;
  __bf16* dst = Eout + ((size_t)b * Lpad + t) * EMB + lane * 4;
  union { __bf16 h[4]; uint2 u; } pk;
  if (t < L) {
    int id = ids[(size_t)b * L + t];
    float4 e = *reinterpret_cast<const float4*>(emb + (size_t)id * EMB + lane * 4);
    float sq = e.x * e.x + e.y * e.y + e.z * e.z + e.w * e.w;
    sq = reduce_wave32(sq);
    float rn = rsqrtf(sq);
    pk.h[0] = (__bf16)(e.x * rn);
    pk.h[1] = (__bf16)(e.y * rn);
    pk.h[2] = (__bf16)(e.z * rn);
    pk.h[3] = (__bf16)(e.w * rn);
  } else {
    pk.u.x = 0u; pk.u.y = 0u;
  }
  *reinterpret_cast<uint2*>(dst) = pk.u;
}

// ---------------- conv(n) + bias + tanh as WMMA GEMM ----------------------
// Ein : (NB, L+2, EMB) bf16 token-major (zero padded rows at end)
// W   : n slices of 128x128 bf16, k-major (slice k = W[:, :, k])
// Yout: (NB, L, NF) bf16 token-major
template <int L, int NTILE>
__global__ __launch_bounds__(256)
void conv_kernel(const __bf16* __restrict__ Ein,
                 const __bf16* __restrict__ W,
                 const float* __restrict__ bias,
                 int n,
                 __bf16* __restrict__ Yout) {
  __shared__ __attribute__((aligned(16))) __bf16 Et[(NTILE + 2) * EMB];
  const int b   = blockIdx.y;
  const int t0  = blockIdx.x * NTILE;
  const int tid = threadIdx.x;

  // stage (NTILE+2) token rows into LDS, 16B chunks
  const int nchunk = (NTILE + 2) * (EMB / 8);
  const __bf16* gsrc = Ein + ((size_t)b * (L + 2) + t0) * EMB;
#ifdef HAVE_ASYNC_LDS
  for (int i = tid; i < nchunk; i += 256) {
    __builtin_amdgcn_global_load_async_to_lds_b128(
        (__attribute__((address_space(1))) v4i_vs*)(gsrc + (size_t)i * 8),
        (__attribute__((address_space(3))) v4i_vs*)(&Et[(size_t)i * 8]),
        0, 0);
  }
  wait_async0();
#else
  {
    const uint4* src = reinterpret_cast<const uint4*>(gsrc);
    uint4* dst = reinterpret_cast<uint4*>(Et);
    for (int i = tid; i < nchunk; i += 256) dst[i] = src[i];
  }
#endif
  __syncthreads();

  const int wave = tid >> 5;
  const int fM   = wave * 16;
  const int lane = tid & 31;
  const int g    = lane >> 4;
  const int nIdx = lane & 15;

  v8f acc[NTILE / 16];
#pragma unroll
  for (int j = 0; j < NTILE / 16; ++j) acc[j] = v8f_zero();

  for (int k = 0; k < n; ++k) {
    const __bf16* Wk = W + (size_t)k * NF * EMB + (size_t)fM * EMB;
#pragma unroll
    for (int c0 = 0; c0 < EMB; c0 += 32) {
      v16bf a = frag_a_rm(Wk + c0, EMB);
#pragma unroll
      for (int j = 0; j < NTILE / 16; ++j) {
        v16bf bf = frag_b_kc(Et + (size_t)(j * 16 + k) * EMB + c0, EMB);
        acc[j] = wmma_bf16(a, bf, acc[j]);
      }
    }
  }

  // epilogue: + bias, tanh, bf16, token-major store (8 contiguous f per lane)
#pragma unroll
  for (int j = 0; j < NTILE / 16; ++j) {
    v8bf pk;
#pragma unroll
    for (int r = 0; r < 8; ++r) {
      int f = fM + 8 * g + r;                 // C layout: M = r + 8*g
      pk[r] = (__bf16)tanhf(acc[j][r] + bias[f]);
    }
    int t = t0 + j * 16 + nIdx;               // C layout: N = lane & 15
    *reinterpret_cast<v8bf*>(
        Yout + ((size_t)b * L + t) * NF + fM + 8 * g) = pk;
  }
}

// ---------------- fused sim GEMM + RBF kernel pooling ---------------------
// qc: (NB, 32, NF) bf16 ; dc: (NB, LD, NF) bf16 ; match: (NB, 3, 32, NKK) f32
// Block handles (b, 256 d-tokens); wave w owns 32 d-tokens x both q tiles.
__global__ __launch_bounds__(256)
void simpool_kernel(const __bf16* __restrict__ qc,
                    const __bf16* __restrict__ dc,
                    float* __restrict__ match,
                    int ngram_idx) {
  __shared__ float sm[LQ * NKK];
  const int b   = blockIdx.y;
  const int t0  = blockIdx.x * 256;
  const int tid = threadIdx.x;
  for (int i = tid; i < LQ * NKK; i += 256) sm[i] = 0.f;
  __syncthreads();

  const int wave = tid >> 5;
  const int lane = tid & 31;
  const int g    = lane >> 4;
  const int nIdx = lane & 15;
  const int tw   = t0 + wave * 32;

  v8f acc[2][2];
#pragma unroll
  for (int mt = 0; mt < 2; ++mt)
#pragma unroll
    for (int nt = 0; nt < 2; ++nt) acc[mt][nt] = v8f_zero();

#pragma unroll
  for (int c0 = 0; c0 < NF; c0 += 32) {
    v16bf a0 = frag_a_rm(qc + ((size_t)b * LQ + 0)  * NF + c0, NF);
    v16bf a1 = frag_a_rm(qc + ((size_t)b * LQ + 16) * NF + c0, NF);
    v16bf b0 = frag_b_kc(dc + ((size_t)b * LD + tw + 0)  * NF + c0, NF);
    v16bf b1 = frag_b_kc(dc + ((size_t)b * LD + tw + 16) * NF + c0, NF);
    acc[0][0] = wmma_bf16(a0, b0, acc[0][0]);
    acc[0][1] = wmma_bf16(a0, b1, acc[0][1]);
    acc[1][0] = wmma_bf16(a1, b0, acc[1][0]);
    acc[1][1] = wmma_bf16(a1, b1, acc[1][1]);
  }

  // RBF kernels in base-2 quadratic form:
  //   exp(-c*(s-mu)^2) = exp2(ak*s^2 + bk*s + dk),  c_lg = c*log2(e)
  // mus: -0.9..0.9 step 0.2 (sigma 0.1) + exact-match mu=1.0 sigma=1e-3
  const float mus[NKK] = {-0.9f, -0.7f, -0.5f, -0.3f, -0.1f,
                           0.1f,  0.3f,  0.5f,  0.7f,  0.9f, 1.0f};
  const float cc[NKK]  = {50.f, 50.f, 50.f, 50.f, 50.f,
                          50.f, 50.f, 50.f, 50.f, 50.f, 500000.f};
  float ak[NKK], bk[NKK], dk[NKK];
#pragma unroll
  for (int k = 0; k < NKK; ++k) {          // fully constant-folded
    float c_lg = cc[k] * 1.44269504f;
    ak[k] = -c_lg;
    bk[k] = 2.f * mus[k] * c_lg;
    dk[k] = -mus[k] * mus[k] * c_lg;
  }

#pragma unroll
  for (int mt = 0; mt < 2; ++mt) {
#pragma unroll
    for (int r = 0; r < 8; ++r) {
      float s0 = acc[mt][0][r];
      float s1 = acc[mt][1][r];
      int qq = mt * 16 + 8 * g + r;        // C layout: M = r + 8*g
#pragma unroll
      for (int k = 0; k < NKK; ++k) {
        float p = fast_exp2(fmaf(fmaf(ak[k], s0, bk[k]), s0, dk[k]))
                + fast_exp2(fmaf(fmaf(ak[k], s1, bk[k]), s1, dk[k]));
        p = reduce_half16(p);              // sum over this wave's 32 d-tokens
        if (nIdx == 0) atomicAdd(&sm[qq * NKK + k], p);
      }
    }
  }
  __syncthreads();
  float* mb = match + ((size_t)b * 3 + ngram_idx) * LQ * NKK;
  for (int i = tid; i < LQ * NKK; i += 256) atomicAdd(&mb[i], sm[i]);
}

// ---------------- final: log1p, sum over q, linear ------------------------
__global__ __launch_bounds__(256)
void final_kernel(const float* __restrict__ match,
                  const float* __restrict__ lw,
                  float* __restrict__ out) {
  __shared__ float red[256];
  const int b = blockIdx.x;
  const int tid = threadIdx.x;
  float s = 0.f;
  for (int i = tid; i < 3 * LQ * NKK; i += 256) {
    int n   = i / (LQ * NKK);
    int rem = i - n * (LQ * NKK);
    int k   = rem % NKK;
    s += lw[n * NKK + k] * log1pf(match[(size_t)b * 3 * LQ * NKK + i]);
  }
  red[tid] = s;
  __syncthreads();
  for (int o = 128; o > 0; o >>= 1) {
    if (tid < o) red[tid] += red[tid + o];
    __syncthreads();
  }
  if (tid == 0) out[b] = red[0];
}

// ---------------- host launcher ----------------
extern "C" void kernel_launch(void* const* d_in, const int* in_sizes, int n_in,
                              void* d_out, int out_size, void* d_ws,
                              size_t ws_size, hipStream_t stream) {
  const int*   q   = (const int*)d_in[0];
  const int*   d   = (const int*)d_in[1];
  const float* emb = (const float*)d_in[4];
  const float* w1  = (const float*)d_in[5];
  const float* b1  = (const float*)d_in[6];
  const float* w2  = (const float*)d_in[7];
  const float* b2  = (const float*)d_in[8];
  const float* w3  = (const float*)d_in[9];
  const float* b3  = (const float*)d_in[10];
  const float* lw  = (const float*)d_in[11];
  float* out = (float*)d_out;

  char* ws = (char*)d_ws;
  size_t off = 0;
  auto carve = [&](size_t bytes) -> char* {
    char* p = ws + off;
    off = (off + bytes + 255) & ~(size_t)255;
    return p;
  };
  __bf16* Eq    = (__bf16*)carve((size_t)NB * (LQ + 2) * EMB * 2);
  __bf16* Ed    = (__bf16*)carve((size_t)NB * (LD + 2) * EMB * 2);
  __bf16* Wbf   = (__bf16*)carve((size_t)6 * NF * EMB * 2);
  __bf16* Qc    = (__bf16*)carve((size_t)NB * LQ * NF * 2);
  __bf16* Dc    = (__bf16*)carve((size_t)NB * LD * NF * 2);
  float*  Match = (float*)carve((size_t)NB * 3 * LQ * NKK * 4);

  // 1. zero the match accumulator
  {
    int n = NB * 3 * LQ * NKK;
    zero_f32_kernel<<<(n + 255) / 256, 256, 0, stream>>>(Match, n);
  }
  // 2. repack conv weights to bf16 k-major slices
  prep_weights_kernel<<<(6 * NF * EMB + 255) / 256, 256, 0, stream>>>(
      w1, w2, w3, Wbf);
  // 3. embed + normalize (q, d)
  {
    int ntq = NB * (LQ + 2);
    embed_kernel<<<(ntq * 32 + 255) / 256, 256, 0, stream>>>(
        q, emb, Eq, LQ, LQ + 2, ntq);
    int ntd = NB * (LD + 2);
    embed_kernel<<<(ntd * 32 + 255) / 256, 256, 0, stream>>>(
        d, emb, Ed, LD, LD + 2, ntd);
  }
  // 4. per-ngram: conv(q), conv(d), fused sim+pool (Qc/Dc reused per ngram)
  const int    slice_start[3] = {0, 1, 3};
  const float* biases[3] = {b1, b2, b3};
  for (int n = 1; n <= 3; ++n) {
    const __bf16* Wn = Wbf + (size_t)slice_start[n - 1] * NF * EMB;
    conv_kernel<LQ, 32><<<dim3(LQ / 32, NB), 256, 0, stream>>>(
        Eq, Wn, biases[n - 1], n, Qc);
    conv_kernel<LD, 64><<<dim3(LD / 64, NB), 256, 0, stream>>>(
        Ed, Wn, biases[n - 1], n, Dc);
    simpool_kernel<<<dim3(LD / 256, NB), 256, 0, stream>>>(
        Qc, Dc, Match, n - 1);
  }
  // 5. final reduction + linear
  final_kernel<<<NB, 256, 0, stream>>>(Match, lw, out);
}
